// classifier_LSTM_per_channel_27668179321059
// MI455X (gfx1250) — matrix operand order repaired and checked by reading, not compiled, not run on hardware
//
#include <hip/hip_runtime.h>
#include <hip/hip_bf16.h>
#include <cstddef>

// ---------------------------------------------------------------------------
// Problem constants
// ---------------------------------------------------------------------------
#define BATCH   512
#define CHANS   128
#define TSTEPS  440
#define HID     128          // H1 == H2 == C == 128
#define GATES   512          // 4*HID
#define OUTDIM  40

typedef _Float16 f16;
typedef __attribute__((ext_vector_type(16))) _Float16 v16h;
typedef __attribute__((ext_vector_type(8)))  _Float16 v8h;
typedef __attribute__((ext_vector_type(4)))  _Float16 v4h;
typedef __attribute__((ext_vector_type(8)))  float    v8f;

// LDS layout (bytes): gate buffer + two activation tiles only (weights live in
// VGPRs).  Activation rows padded to 136 f16 (272 B = 17*16 -> every b128
// sub-load stays 16-B aligned while breaking the power-of-two bank stride).
#define APITCH      136
#define OFF_GBUF    0
#define OFF_XT      (16 * GATES * 4)                      // 32768
#define OFF_HT      (OFF_XT + 16 * APITCH * 2)            // 37120
#define SMEM_BYTES  (OFF_HT + 16 * APITCH * 2)            // 41472

__device__ __forceinline__ float sigf(float x) {
    return 1.0f / (1.0f + __expf(-x));
}
__device__ __forceinline__ float tanh_fast(float x) {
    x = fminf(15.0f, fmaxf(-15.0f, x));
    float e = __expf(2.0f * x);
    return (e - 1.0f) / (e + 1.0f);
}

// Build a v16h A-operand from two 16-byte LDS chunks.
__device__ __forceinline__ v16h load_pair(const f16* p0, const f16* p1) {
    union { v16h v; v8h h[2]; } u;
    u.h[0] = *(const v8h*)p0;
    u.h[1] = *(const v8h*)p1;
    return u.v;
}

// Convert 16 contiguous f32 (global) into a v16h B-operand fragment.
__device__ __forceinline__ v16h cvt_row16(const float* __restrict__ p) {
    v16h r;
#pragma unroll
    for (int j = 0; j < 16; ++j) r[j] = (f16)p[j];
    return r;
}

__device__ __forceinline__ v8f wmma_f16(v16h a, v16h b, v8f c) {
    // D = A(16x32 f16) * B(32x16 f16) + C(16x16 f32)
    return __builtin_amdgcn_wmma_f32_16x16x32_f16(
        /*neg_a=*/false, a, /*neg_b=*/false, b,
        /*c_mod=*/(short)0, c, /*reuse_a=*/false, /*reuse_b=*/false);
}

// ---------------------------------------------------------------------------
// Kernel 1: per-channel hidden-size-1 LSTM (scalar recurrence, memory bound).
// X[B][C][T] f32  ->  feats[B][T][C] f16  (h is bounded by tanh -> f16 safe)
// ---------------------------------------------------------------------------
__global__ void perchan_lstm_kernel(const float* __restrict__ X,
                                    const float* __restrict__ Wih0,
                                    const float* __restrict__ Whh0,
                                    const float* __restrict__ b0,
                                    f16* __restrict__ feats) {
    int s  = blockIdx.x * blockDim.x + threadIdx.x;   // 0 .. B*C-1
    int b  = s >> 7;
    int ch = s & 127;
    const float* xp = X + (size_t)s * TSTEPS;                  // contiguous in t
    f16* fp = feats + (size_t)b * TSTEPS * CHANS + ch;         // stride C in t

    float wi0 = Wih0[0], wi1 = Wih0[1], wi2 = Wih0[2], wi3 = Wih0[3];
    float wh0 = Whh0[0], wh1 = Whh0[1], wh2 = Whh0[2], wh3 = Whh0[3];
    float bb0 = b0[0],   bb1 = b0[1],   bb2 = b0[2],   bb3 = b0[3];

    float h = 0.0f, c = 0.0f;
    for (int t = 0; t < TSTEPS; t += 4) {
        float4 x4 = *(const float4*)(xp + t);     // 1760-B row pitch -> aligned
        float xs[4] = { x4.x, x4.y, x4.z, x4.w };
#pragma unroll
        for (int q = 0; q < 4; ++q) {
            float x  = xs[q];
            float ig = sigf(wi0 * x + wh0 * h + bb0);
            float fg = sigf(wi1 * x + wh1 * h + bb1);
            float gg = tanh_fast(wi2 * x + wh2 * h + bb2);
            float og = sigf(wi3 * x + wh3 * h + bb3);
            c = fg * c + ig * gg;
            h = og * tanh_fast(c);
            fp[(size_t)(t + q) * CHANS] = (f16)h;
        }
    }
}

// ---------------------------------------------------------------------------
// Kernel 2: fused WMMA LSTM layer.  Each workgroup owns 16 batch rows and
// walks all 440 timesteps.  Weight B-operands (f16) are loop-invariant and
// live entirely in VGPRs (128 VGPRs/lane); LDS only holds the gate buffer and
// the x/h activation tiles.  Per step per wave: 16 v_wmma_f32_16x16x32_f16.
// FINAL=true additionally applies the classifier head + softmax at t=T-1.
// ---------------------------------------------------------------------------
template <bool FINAL>
__global__ __launch_bounds__(512, 1)
void lstm_layer_kernel(const f16* __restrict__ xin,     // [B][T][128] f16
                       const float* __restrict__ Wih,   // [512][128] f32
                       const float* __restrict__ Whh,   // [512][128] f32
                       const float* __restrict__ bias,  // [512] f32
                       f16* __restrict__ hout,          // [B][T][128] f16 (layer1)
                       const float* __restrict__ Wlin,  // [40][128] f32
                       const float* __restrict__ blin,  // [40] f32
                       float* __restrict__ out) {       // [B][40] f32
    extern __shared__ char smem[];
    float* gbuf = (float*)(smem + OFF_GBUF);    // [16][512] gate pre-activations
    f16*   x_l  = (f16*)(smem + OFF_XT);        // [16][APITCH] input tile
    f16*   h_l  = (f16*)(smem + OFF_HT);        // [16][APITCH] hidden state

    const int tid  = threadIdx.x;               // 512 threads = 16 waves
    const int lane = tid & 31;
    const int wv   = tid >> 5;
    const int b0r  = blockIdx.x * 16;

    const int mrow   = lane & 15;                       // A-operand row
    const int ncol0  = wv * 32 + (lane & 15);           // tile0 column for this lane
    const int koff_a = (lane < 16) ? 0 : 16;            // A layout lane split
    const int koff_b = (lane < 16) ? 0 : 16;            // B layout lane split
    const int mtop   = (lane >> 4) * 8;                 // C/D row base
    const float bias0 = bias[ncol0];
    const float bias1 = bias[ncol0 + 16];

    // --- one-time: weight B-operands f32(global) -> f16(VGPRs) -------------
    // wave wv owns N-tiles {2wv, 2wv+1}; lane holds column n = tile*16+lane%16
    v16h bih[2][4], bhh[2][4];
#pragma unroll
    for (int tile = 0; tile < 2; ++tile) {
        const int n = ncol0 + tile * 16;
#pragma unroll
        for (int kc = 0; kc < 4; ++kc) {
            const int kb = kc * 32 + koff_b;
            bih[tile][kc] = cvt_row16(Wih + (size_t)n * HID + kb);
            bhh[tile][kc] = cvt_row16(Whh + (size_t)n * HID + kb);
        }
    }

    for (int idx = tid; idx < 16 * APITCH; idx += 512) h_l[idx] = (f16)0.0f;
    __syncthreads();

    // per-thread cell state: thread owns units (m = tid/32, j = lane*4 .. +3)
    float c_state[4] = {0.0f, 0.0f, 0.0f, 0.0f};
    const int mld = tid >> 5;                           // row this thread stages

    const f16* arow_x = x_l + mrow * APITCH;
    const f16* arow_h = h_l + mrow * APITCH;

    for (int t = 0; t < TSTEPS; ++t) {
        // ---- stage x_t tile [16][128] f16 into LDS (8 B per thread) ----
        {
            const f16* src = xin + ((size_t)(b0r + mld) * TSTEPS + t) * HID;
            *(v4h*)(x_l + mld * APITCH + lane * 4) = *((const v4h*)src + lane);
            int tn = (t + 1 < TSTEPS) ? t + 1 : t;      // prefetch next step
            __builtin_prefetch(xin + ((size_t)(b0r + mld) * TSTEPS + tn) * HID, 0, 0);
        }
        __syncthreads();

        // ---- GEMM phase: g = x_t*Wih^T + h*Whh^T  (two 16x16 tiles / wave) ----
        v8f acc0 = {}; v8f acc1 = {};
#pragma unroll
        for (int kc = 0; kc < 4; ++kc) {
            const int ka = kc * 32 + koff_a;
            v16h ax = load_pair(arow_x + ka, arow_x + ka + 8);
            v16h ah = load_pair(arow_h + ka, arow_h + ka + 8);
            acc0 = wmma_f16(ax, bih[0][kc], acc0);
            acc0 = wmma_f16(ah, bhh[0][kc], acc0);
            acc1 = wmma_f16(ax, bih[1][kc], acc1);
            acc1 = wmma_f16(ah, bhh[1][kc], acc1);
        }
        // scatter to gate buffer per C/D layout: m = v + 8*(lane/16), n = ncol0
#pragma unroll
        for (int v = 0; v < 8; ++v) {
            gbuf[(mtop + v) * GATES + ncol0]      = acc0[v] + bias0;
            gbuf[(mtop + v) * GATES + ncol0 + 16] = acc1[v] + bias1;
        }
        __syncthreads();

        // ---- elementwise gate phase: 4 hidden units per thread ----
        {
            const int j = lane * 4;
            const float* gr = gbuf + mld * GATES;
            float hv[4];
#pragma unroll
            for (int q = 0; q < 4; ++q) {
                int jj   = j + q;
                float ig = gr[jj];
                float fg = gr[128 + jj];
                float gg = gr[256 + jj];
                float og = gr[384 + jj];
                float cc = sigf(fg) * c_state[q] + sigf(ig) * tanh_fast(gg);
                c_state[q] = cc;
                hv[q] = sigf(og) * tanh_fast(cc);
            }
            v4h hp = { (f16)hv[0], (f16)hv[1], (f16)hv[2], (f16)hv[3] };
            *(v4h*)(h_l + mld * APITCH + j) = hp;
            if (!FINAL)
                *(v4h*)(hout + ((size_t)(b0r + mld) * TSTEPS + t) * HID + j) = hp;
        }
        __syncthreads();
    }

    if (FINAL) {
        // classifier head on h_T (in h_l), reusing gbuf LDS region
        float* wl = gbuf;                 // 40*128 f32 = 20 KB (gbuf is 32 KB)
        float* bl = (float*)x_l;          // 40 f32 (x tile no longer needed)
        float* lg = bl + 64;              // 16*40 logits
        for (int idx = tid; idx < OUTDIM * HID; idx += 512) wl[idx] = Wlin[idx];
        if (tid < OUTDIM) bl[tid] = blin[tid];
        __syncthreads();
        for (int idx = tid; idx < 16 * OUTDIM; idx += 512) {
            int m = idx / OUTDIM, o = idx % OUTDIM;
            float s = bl[o];
            const f16* hr = h_l + m * APITCH;
            const float* wr = wl + o * HID;
#pragma unroll 8
            for (int k = 0; k < HID; ++k) s += (float)hr[k] * wr[k];
            lg[m * OUTDIM + o] = s;
        }
        __syncthreads();
        if (tid < 16) {
            const float* row = lg + tid * OUTDIM;
            float mx = -1e30f;
            for (int o = 0; o < OUTDIM; ++o) mx = fmaxf(mx, row[o]);
            float s = 0.0f;
            for (int o = 0; o < OUTDIM; ++o) s += __expf(row[o] - mx);
            float inv = 1.0f / s;
            float* dst = out + (size_t)(b0r + tid) * OUTDIM;
            for (int o = 0; o < OUTDIM; ++o) dst[o] = __expf(row[o] - mx) * inv;
        }
    }
}

// ---------------------------------------------------------------------------
// Host-side launcher
// ---------------------------------------------------------------------------
extern "C" void kernel_launch(void* const* d_in, const int* in_sizes, int n_in,
                              void* d_out, int out_size, void* d_ws, size_t ws_size,
                              hipStream_t stream) {
    (void)in_sizes; (void)n_in; (void)out_size; (void)ws_size;

    const float* X    = (const float*)d_in[0];
    const float* Wih0 = (const float*)d_in[1];
    const float* Whh0 = (const float*)d_in[2];
    const float* b0   = (const float*)d_in[3];
    const float* Wih1 = (const float*)d_in[4];
    const float* Whh1 = (const float*)d_in[5];
    const float* b1   = (const float*)d_in[6];
    const float* Wih2 = (const float*)d_in[7];
    const float* Whh2 = (const float*)d_in[8];
    const float* b2   = (const float*)d_in[9];
    const float* Wlin = (const float*)d_in[10];
    const float* blin = (const float*)d_in[11];
    float* out = (float*)d_out;

    // workspace: feats [B][T][C] f16, then h1 [B][T][H1] f16
    const size_t actBytes = (size_t)BATCH * TSTEPS * CHANS * sizeof(f16); // 57.7 MB
    f16* feats = (f16*)d_ws;
    f16* h1buf = (f16*)((char*)d_ws + actBytes);

    // Stage 1: 65536 scalar LSTMs
    perchan_lstm_kernel<<<BATCH * CHANS / 256, 256, 0, stream>>>(X, Wih0, Whh0, b0, feats);

    // Stage 2: lstm_full1 (writes h1 sequence)
    lstm_layer_kernel<false><<<BATCH / 16, 512, SMEM_BYTES, stream>>>(
        feats, Wih1, Whh1, b1, h1buf, nullptr, nullptr, nullptr);

    // Stage 3: lstm_full2 + linear + softmax
    lstm_layer_kernel<true><<<BATCH / 16, 512, SMEM_BYTES, stream>>>(
        h1buf, Wih2, Whh2, b2, nullptr, Wlin, blin, out);
}
